// PsdQuadratic_40200893891172
// MI455X (gfx1250) — matrix operand order repaired and checked by reading, not compiled, not run on hardware
//
#include <hip/hip_runtime.h>

// ============================================================================
// PsdQuadratic on MI455X (gfx1250): fp16x3-split WMMA implementation.
//
// y_i = || x_i^T tril(L) ||^2 + b.x_i + c
//
// x@Lt (17.2 GFLOP fp32) is executed as three f16 WMMA products
// (hi*hi + lo*hi + hi*lo) on V_WMMA_F32_16X16X32_F16 with f32 accumulation.
// RTZ pack (v_cvt_pk_rtz_f16_f32) makes the residual exact, so the pair
// (hi,lo) carries ~22 mantissa bits => ~fp32 accuracy, while running at the
// 16-bit matrix rate. 3x WMMA work (~51.5 GFLOP) lands at the same ~10-12us
// as the 268MB/23.3TB/s HBM floor -> balanced roofline.
// Register budget kept < 256 VGPRs (no spills, no VGPR-MSB switching):
// A fragments 64 + acc 16 + B 16 + temps ~30.
// ============================================================================

#define D_DIM        128
#define TILE_M       16
#define BLOCK_THREADS 256
#define GRID_BLOCKS  512

typedef __attribute__((ext_vector_type(16))) _Float16     v16h;
typedef __attribute__((ext_vector_type(2)))  __fp16       pkh2;  // cvt_pkrtz result
typedef __attribute__((ext_vector_type(8)))  float        v8f;
typedef __attribute__((ext_vector_type(8)))  unsigned int v8u;
typedef __attribute__((ext_vector_type(4)))  float        v4f;

__global__ __launch_bounds__(BLOCK_THREADS)
void psd_quadratic_f16x3(const float* __restrict__ xg,
                         const float* __restrict__ Lg,
                         const float* __restrict__ bg,
                         const float* __restrict__ cg,
                         float* __restrict__ out,
                         int ntiles) {
  // tril(L) split into f16 hi/lo, stored pre-swizzled in the WMMA
  // B-fragment (32x16, K x N) lane layout:
  //   lane = (n&15) | ((k&8)<<1), element = ((k&16)>>1) | (k&7)
  // so a lane's fragment is 32 contiguous bytes -> two ds_load_b128.
  __shared__ __align__(32) unsigned short sLhi[D_DIM * D_DIM];  // 32 KB
  __shared__ __align__(32) unsigned short sLlo[D_DIM * D_DIM];  // 32 KB

  const int tid = threadIdx.x;

  // ---- One-time per-block prep: tril + f16 split + swizzle into LDS ----
  for (int idx = tid; idx < D_DIM * D_DIM; idx += BLOCK_THREADS) {
    const int k = idx >> 7;            // K index (row of Lt)
    const int n = idx & (D_DIM - 1);   // N index (col of Lt)
    const float val = (n <= k) ? Lg[idx] : 0.0f;   // lower-triangular
    const _Float16 hi = (_Float16)val;             // RNE
    const _Float16 lo = (_Float16)(val - (float)hi);
    const int kc = k >> 5;
    const int nt = n >> 4;
    const int ln = (n & 15) | (((k >> 3) & 1) << 4);
    const int off = ((kc * 8 + nt) * 32 + ln) * 16 + (((k >> 4) & 1) << 3) + (k & 7);
    sLhi[off] = __builtin_bit_cast(unsigned short, hi);
    sLlo[off] = __builtin_bit_cast(unsigned short, lo);
  }
  __syncthreads();

  const int lane   = tid & 31;
  const int lmod   = lane & 15;
  const int laneHi = lane >> 4;
  const int waveId = (blockIdx.x * BLOCK_THREADS + tid) >> 5;
  const int nwaves = (gridDim.x * BLOCK_THREADS) >> 5;
  const float cval = cg[0];

#pragma unroll 1
  for (int tile = waveId; tile < ntiles; tile += nwaves) {  // wave-uniform
    // Lane pair (l, l+16) shares row lmod of the 16-row tile.
    const float* __restrict__ xrow = xg + (size_t)(tile * TILE_M + lmod) * D_DIM;

    // ---- Phase 1: load x once, split to f16 hi/lo A-fragments, fold b.x ----
    // A-fragment (16x32 f16) ISA layout: this lane holds row lmod,
    // K = k0..k0+7 (vgpr0-3) and k0+16..k0+23 (vgpr4-7), k0 = kc*32 + 8*laneHi.
    v16h Ah[4], Al[4];
    float bdot = 0.0f;
#pragma unroll
    for (int kc = 0; kc < 4; ++kc) {
      const int k0 = kc * 32 + laneHi * 8;
      v4f xv4[4], bv4[4];
      xv4[0] = __builtin_nontemporal_load((const v4f*)(xrow + k0));
      xv4[1] = __builtin_nontemporal_load((const v4f*)(xrow + k0 + 4));
      xv4[2] = __builtin_nontemporal_load((const v4f*)(xrow + k0 + 16));
      xv4[3] = __builtin_nontemporal_load((const v4f*)(xrow + k0 + 20));
      bv4[0] = *(const v4f*)(bg + k0);
      bv4[1] = *(const v4f*)(bg + k0 + 4);
      bv4[2] = *(const v4f*)(bg + k0 + 16);
      bv4[3] = *(const v4f*)(bg + k0 + 20);

      v8u hv, lv;
#pragma unroll
      for (int q = 0; q < 4; ++q) {
#pragma unroll
        for (int r = 0; r < 2; ++r) {
          const float a = xv4[q][2 * r];
          const float b = xv4[q][2 * r + 1];
          // 1 op: pack two f32 -> f16 pair with round-toward-zero
          const pkh2 ph = __builtin_amdgcn_cvt_pkrtz(a, b);
          // residual is exact in f32 (hi holds the top mantissa bits)
          const pkh2 pl = __builtin_amdgcn_cvt_pkrtz(a - (float)ph[0],
                                                     b - (float)ph[1]);
          hv[2 * q + r] = __builtin_bit_cast(unsigned int, ph);
          lv[2 * q + r] = __builtin_bit_cast(unsigned int, pl);
        }
        bdot = __builtin_fmaf(xv4[q][0], bv4[q][0],
               __builtin_fmaf(xv4[q][1], bv4[q][1],
               __builtin_fmaf(xv4[q][2], bv4[q][2],
               __builtin_fmaf(xv4[q][3], bv4[q][3], bdot))));
      }
      Ah[kc] = __builtin_bit_cast(v16h, hv);
      Al[kc] = __builtin_bit_cast(v16h, lv);
    }

    // ---- Phase 2: 96 WMMAs, 8 N-tiles x 4 K-chunks x 3 products ----
    float p[8];
#pragma unroll
    for (int j = 0; j < 8; ++j) p[j] = 0.0f;

#pragma unroll 2
    for (int nt = 0; nt < 8; ++nt) {
      v8f a0 = (v8f){0, 0, 0, 0, 0, 0, 0, 0};   // hi*hi chain
      v8f a1 = (v8f){0, 0, 0, 0, 0, 0, 0, 0};   // correction chains
#pragma unroll
      for (int kc = 0; kc < 4; ++kc) {
        const int boff = ((kc * 8 + nt) * 32 + lane) * 16;  // halfs; 32B aligned
        const v16h Bhi = *(const v16h*)(sLhi + boff);
        const v16h Blo = *(const v16h*)(sLlo + boff);
        a0 = __builtin_amdgcn_wmma_f32_16x16x32_f16(
            false, Ah[kc], false, Bhi, (short)0, a0, false, false);
        a1 = __builtin_amdgcn_wmma_f32_16x16x32_f16(
            false, Al[kc], false, Bhi, (short)0, a1, false, false);
        a1 = __builtin_amdgcn_wmma_f32_16x16x32_f16(
            false, Ah[kc], false, Blo, (short)0, a1, false, false);
      }
      // C layout: lane = col N (per 16-lane half), vgpr j = row M (+8 upper).
#pragma unroll
      for (int j = 0; j < 8; ++j) {
        const float t = a0[j] + a1[j];
        p[j] = __builtin_fmaf(t, t, p[j]);
      }
    }

    // ---- Phase 3: reduce across the 16 columns (xor stays within halves) ----
#pragma unroll
    for (int m = 1; m <= 8; m <<= 1) {
#pragma unroll
      for (int j = 0; j < 8; ++j) p[j] += __shfl_xor(p[j], m, 32);
    }
    // b.x: lanes l and l+16 each hold half of row lmod's dot product.
    const float bd = bdot + __shfl_xor(bdot, 16, 32);

    // Route row sums so lane r (0..15) holds row r: every lane selects its
    // t[lane&7]; lower half covers M=0..7, upper half M=8..15 via one shfl.
    float sel = p[0];
#pragma unroll
    for (int j = 1; j < 8; ++j) sel = ((lane & 7) == j) ? p[j] : sel;
    const float up = __shfl(sel, lmod + 16, 32);

    if (lane < 16) {
      const float y = ((lmod < 8) ? sel : up) + bd + cval;
      __builtin_nontemporal_store(y, out + tile * TILE_M + lane);
    }
  }
}

extern "C" void kernel_launch(void* const* d_in, const int* in_sizes, int n_in,
                              void* d_out, int out_size, void* d_ws, size_t ws_size,
                              hipStream_t stream) {
  const float* x = (const float*)d_in[0];   // [B, 128] fp32
  const float* L = (const float*)d_in[1];   // [128, 128] fp32
  const float* b = (const float*)d_in[2];   // [128] fp32
  const float* c = (const float*)d_in[3];   // [1] fp32
  float* out = (float*)d_out;               // [B] fp32

  const int batch  = in_sizes[0] / D_DIM;   // 524288
  const int ntiles = batch / TILE_M;        // 32768 (exact)

  hipLaunchKernelGGL(psd_quadratic_f16x3,
                     dim3(GRID_BLOCKS), dim3(BLOCK_THREADS), 0, stream,
                     x, L, b, c, out, ntiles);
}
